// RNNModel_25108378812705
// MI455X (gfx1250) — matrix operand order repaired and checked by reading, not compiled, not run on hardware
//
#include <hip/hip_runtime.h>
#include <cstdint>

typedef __attribute__((ext_vector_type(16))) _Float16 v16h;
typedef __attribute__((ext_vector_type(8)))  float    v8f;
typedef unsigned int v4u __attribute__((ext_vector_type(4)));
typedef int          v8i __attribute__((ext_vector_type(8)));
typedef int          v4i __attribute__((ext_vector_type(4)));

#define VOCAB  80
#define HID    128
#define BATCH  256
#define TLEN   512
#define ROWS   16                    // batch rows per block
#define TCHUNK 32                    // token staging chunk (double-buffered)
#define NCHUNK (TLEN / TCHUNK)       // 16

// Branch-free tanh: tanh(x) = 1 - 2/(exp(2x)+1), native exp2/rcp.
// Saturates correctly: exp2->inf => 1.0 ; exp2->0 => -1.0.
__device__ __forceinline__ float fast_tanh(float x) {
    float e = __builtin_amdgcn_exp2f(x * 2.88539008177792681f);  // exp(2x)
    return 1.0f - 2.0f * __builtin_amdgcn_rcpf(e + 1.0f);
}

// 32-bit LDS byte offset of a __shared__ object (generic shared addr low bits).
__device__ __forceinline__ uint32_t lds_off(const void* p) {
    return (uint32_t)(uintptr_t)p;
}

// Issue one TDM 2D tile load (Global -> LDS).  ds_code: 0=1B,1=2B,2=4B,3=8B.
// 1D copies: pass tile_y = 0 (tile_dim1 unused).
__device__ __forceinline__ void tdm_load_2d(uint32_t ldsoff, const void* gptr,
                                            uint32_t tile_x, uint32_t tile_y,
                                            uint32_t tensor_x, uint32_t tensor_y,
                                            uint32_t stride_x, uint32_t ds_code) {
    uint64_t ga = (uint64_t)(uintptr_t)gptr;
    v4u g0;
    g0[0] = 1u;                                                  // count=1, user mode
    g0[1] = ldsoff;                                              // lds_addr
    g0[2] = (uint32_t)ga;                                        // global_addr[31:0]
    g0[3] = (uint32_t)((ga >> 32) & 0x1FFFFFFu) | (2u << 30);    // addr[56:32] | type=2
    v8i g1;
    g1[0] = (int)(ds_code << 16);                                // wg_mask=0, data_size
    g1[1] = (int)((tensor_x & 0xFFFFu) << 16);                   // tensor_dim0[15:0]
    g1[2] = (int)(((tensor_x >> 16) & 0xFFFFu) |
                  ((tensor_y & 0xFFFFu) << 16));                 // dim0[31:16] | dim1[15:0]
    g1[3] = (int)(((tensor_y >> 16) & 0xFFFFu) |
                  ((tile_x & 0xFFFFu) << 16));                   // dim1[31:16] | tile_dim0
    g1[4] = (int)(tile_y & 0xFFFFu);                             // tile_dim1 (tile_dim2=0)
    g1[5] = (int)stride_x;                                       // tensor_dim0_stride[31:0]
    g1[6] = 0;
    g1[7] = 0;
    v4i z = {0, 0, 0, 0};
#if defined(__clang_major__) && (__clang_major__ >= 23)
    v8i z8 = {0, 0, 0, 0, 0, 0, 0, 0};
    __builtin_amdgcn_tensor_load_to_lds(g0, g1, z, z, z8, 0);
#else
    __builtin_amdgcn_tensor_load_to_lds(g0, g1, z, z, 0);
#endif
}

// ---------------------------------------------------------------------------
// Kernel 0: embW[v][h] = dot(emb[v,:], W_ih[h,:]) + b_ih[h] + b_hh[h]
// (both biases folded into the lookup table; 80x128 f32)
// ---------------------------------------------------------------------------
__global__ void embw_kernel(const float* __restrict__ emb,
                            const float* __restrict__ W_ih,
                            const float* __restrict__ b_ih,
                            const float* __restrict__ b_hh,
                            float* __restrict__ embW) {
    int v = blockIdx.x;
    int h = threadIdx.x;
    const float* er = emb  + v * HID;
    const float* wr = W_ih + h * HID;
    float s = b_ih[h] + b_hh[h];
#pragma unroll 8
    for (int k = 0; k < HID; ++k) s += er[k] * wr[k];
    embW[v * HID + h] = s;
}

// ---------------------------------------------------------------------------
// Kernel 1: recurrence.  One block = 16 batch rows for all 512 steps.
//   h_{t+1} = tanh(embW[x_t] + h_t @ W_hh^T)   via v_wmma_f32_16x16x32_f16
// ---------------------------------------------------------------------------
__global__ __launch_bounds__(256) void rnn_kernel(
    const int*   __restrict__ x,
    const float* __restrict__ W_hh,
    const float* __restrict__ fc_W,
    const float* __restrict__ fc_b,
    const float* __restrict__ embW,
    float*       __restrict__ out)
{
    __shared__ float                  s_embW[VOCAB * HID];          // 40 KB
    __shared__ __align__(32) _Float16 s_hA[2][4][32][16];           //  8 KB dbl-buffered A-layout h
    __shared__ int                    s_tok[2][ROWS][TCHUNK];       //  4 KB dbl-buffered tokens
    __shared__ float                  s_hF[ROWS][HID];              //  8 KB

    const int tid  = threadIdx.x;
    const int lane = tid & 31;
    const int wave = tid >> 5;          // 8 waves; wave w owns cols [16w,16w+16)
    const int hi   = lane >> 4;
    const int nloc = lane & 15;
    const int ncol = wave * 16 + nloc;
    const int row0 = blockIdx.x * ROWS;

    // ---- TDM: stage embW table (1D) + token chunk 0; zero h0 meanwhile ----
    if (wave == 0) {
        tdm_load_2d(lds_off(s_embW), embW,
                    VOCAB * HID, 0, VOCAB * HID, 1, VOCAB * HID, 2);
        tdm_load_2d(lds_off(s_tok[0]), x + row0 * TLEN,
                    TCHUNK, ROWS, TLEN, BATCH, TLEN, 2);
    }
    {
        _Float16* p = &s_hA[0][0][0][0];
        for (int i = tid; i < 2 * 4 * 32 * 16; i += 256) p[i] = (_Float16)0.f;
    }

    // ---- B fragments: W_hh^T columns, resident in VGPRs for all 512 steps ----
    // B layout (32x16, 16-bit): lanes 0-15 hold K=0..7,16..23 of column n, lanes 16-31 K=8..15,24..31.
    v16h bfrag[4];
    {
        const float* wrow = W_hh + ncol * HID;          // B[k][n] = W_hh[n][k]
#pragma unroll
        for (int kc = 0; kc < 4; ++kc) {
            const float* src = wrow + kc * 32 + hi * 8;
            v16h tmp;
#pragma unroll
            for (int j = 0; j < 16; ++j) tmp[j] = (_Float16)src[j < 8 ? j : j + 8];
            bfrag[kc] = tmp;
        }
    }

    // ---- f16 scatter indices for writing h_new into the A layout ----
    // A element (m,k): kc=k>>5, koff=k&31, sub=koff>>3;
    //   lane' = m + 16*(sub&1); vgpr' = 4*(sub>>1) + ((koff&7)>>1); half = koff&1
    int stIdx[8];
    {
        const int k    = ncol;
        const int kc   = k >> 5;
        const int koff = k & 31;
        const int sub  = koff >> 3;
        const int vv   = 4 * (sub >> 1) + ((koff & 7) >> 1);
        const int half = koff & 1;
#pragma unroll
        for (int r = 0; r < 8; ++r) {
            int m  = r + 8 * hi;
            int lp = m + ((sub & 1) ? 16 : 0);
            stIdx[r] = ((kc * 32 + lp) * 16) + vv * 2 + half;
        }
    }

    float hn[8];
    int buf = 0;
    for (int c = 0; c < NCHUNK; ++c) {
        // entry: chunk c's token TDM (issued one chunk early) must be complete
        if (wave == 0) __builtin_amdgcn_s_wait_tensorcnt(0);
        __syncthreads();
        // prefetch chunk c+1's tokens asynchronously into the other buffer
        if (wave == 0 && (c + 1) < NCHUNK) {
            tdm_load_2d(lds_off(s_tok[(c + 1) & 1]),
                        x + row0 * TLEN + (c + 1) * TCHUNK,
                        TCHUNK, ROWS, TLEN, BATCH, TLEN, 2);
        }
        const int tb = c & 1;

        // software-pipelined xi gather: pref holds step tt's accumulator
        v8f pref;
#pragma unroll
        for (int r = 0; r < 8; ++r) {
            int m   = r + 8 * hi;
            int tok = s_tok[tb][m][0];
            pref[r] = s_embW[tok * HID + ncol];
        }

        for (int tt = 0; tt < TCHUNK; ++tt) {
            v8f acc = pref;

            // A fragments (h_t)
            v16h a0 = *(const v16h*)&s_hA[buf][0][lane][0];
            v16h a1 = *(const v16h*)&s_hA[buf][1][lane][0];
            v16h a2 = *(const v16h*)&s_hA[buf][2][lane][0];
            v16h a3 = *(const v16h*)&s_hA[buf][3][lane][0];

            // prefetch next step's gather; hidden under the WMMA chain + tanh
            if (tt + 1 < TCHUNK) {
#pragma unroll
                for (int r = 0; r < 8; ++r) {
                    int m   = r + 8 * hi;
                    int tok = s_tok[tb][m][tt + 1];
                    pref[r] = s_embW[tok * HID + ncol];
                }
            }

            // force all DS loads to issue before the WMMA chain
            __builtin_amdgcn_sched_barrier(0);

            acc = __builtin_amdgcn_wmma_f32_16x16x32_f16(false, a0, false, bfrag[0], (short)0, acc, false, false);
            acc = __builtin_amdgcn_wmma_f32_16x16x32_f16(false, a1, false, bfrag[1], (short)0, acc, false, false);
            acc = __builtin_amdgcn_wmma_f32_16x16x32_f16(false, a2, false, bfrag[2], (short)0, acc, false, false);
            acc = __builtin_amdgcn_wmma_f32_16x16x32_f16(false, a3, false, bfrag[3], (short)0, acc, false, false);

            // branch-free tanh, scatter f16 h_{t+1} into the other A buffer
            _Float16* dst = &s_hA[buf ^ 1][0][0][0];
#pragma unroll
            for (int r = 0; r < 8; ++r) {
                hn[r] = fast_tanh(acc[r]);
                dst[stIdx[r]] = (_Float16)hn[r];
            }
            __syncthreads();   // publish h_{t+1}; order reads(t) before writes(t+1)
            buf ^= 1;
        }
    }

    // ---- final head: out = h_last @ fc_W^T + fc_b  (f32 h, pre-quantization) ----
#pragma unroll
    for (int r = 0; r < 8; ++r) {
        int m = r + 8 * hi;
        s_hF[m][ncol] = hn[r];
    }
    __syncthreads();
    for (int i = tid; i < ROWS * VOCAB; i += 256) {
        int rr = i / VOCAB, v = i % VOCAB;
        const float* w = fc_W + v * HID;
        float s = fc_b[v];
#pragma unroll 8
        for (int k = 0; k < HID; ++k) s += s_hF[rr][k] * w[k];
        out[(row0 + rr) * VOCAB + v] = s;
    }
}

// ---------------------------------------------------------------------------
extern "C" void kernel_launch(void* const* d_in, const int* in_sizes, int n_in,
                              void* d_out, int out_size, void* d_ws, size_t ws_size,
                              hipStream_t stream) {
    const int*   x     = (const int*)  d_in[0];
    const float* emb   = (const float*)d_in[1];
    const float* W_ih  = (const float*)d_in[2];
    const float* W_hh  = (const float*)d_in[3];
    const float* b_ih  = (const float*)d_in[4];
    const float* b_hh  = (const float*)d_in[5];
    const float* fc_W  = (const float*)d_in[6];
    const float* fc_b  = (const float*)d_in[7];
    float* out  = (float*)d_out;
    float* embW = (float*)d_ws;        // 80*128 f32 = 40 KB scratch

    embw_kernel<<<VOCAB, HID, 0, stream>>>(emb, W_ih, b_ih, b_hh, embW);
    rnn_kernel<<<BATCH / ROWS, 256, 0, stream>>>(x, W_hh, fc_W, fc_b, embW, out);
}
